// QuantumNeuralNetworkCL_8727373545716
// MI455X (gfx1250) — compile-verified, matrix operand order checked by reading, not run on hardware
//
#include <hip/hip_runtime.h>

// ---------------------------------------------------------------------------
// Problem constants
// ---------------------------------------------------------------------------
#define NQ      12
#define DIM     4096          // 2^12
#define NPAIRS  2048          // DIM/2
#define BATCH   2048
#define NEXP    36            // 12 qubits x {X,Y,Z}
#define KPAD    64            // padded K for WMMA (f16, 2 chunks of 32)
#define NCLS    5

// ---------------------------------------------------------------------------
// Compile-time gate table (matches reference parameter consumption order)
// kind: 0=U3, 1=U3^dagger (reuses params at p), 2=RZ, 3=RY, 4=CNOT(a=ctrl,b=tgt)
// ---------------------------------------------------------------------------
struct Op { unsigned char kind, a, b; unsigned short p; };
#define NOPS 360
struct OpTable { Op ops[NOPS]; };

struct Builder {
    OpTable T {};
    int n = 0, p = 0;
    constexpr void u3(int w)            { T.ops[n] = Op{0,(unsigned char)w,0,(unsigned short)p}; n++; p += 3; }
    constexpr void u3dag(int w, int pp) { T.ops[n] = Op{1,(unsigned char)w,0,(unsigned short)pp}; n++; }
    constexpr void rz(int w)            { T.ops[n] = Op{2,(unsigned char)w,0,(unsigned short)p}; n++; p += 1; }
    constexpr void ry(int w)            { T.ops[n] = Op{3,(unsigned char)w,0,(unsigned short)p}; n++; p += 1; }
    constexpr void cnot(int c, int t)   { T.ops[n] = Op{4,(unsigned char)c,(unsigned char)t,0}; n++; }
};

constexpr Builder make_builder() {
    Builder B{};
    const int UP[12][2] = {{1,2},{3,4},{5,6},{7,8},{9,10},{11,0},
                           {0,1},{2,3},{4,5},{6,7},{8,9},{10,11}};
    const int VP[6][2]  = {{0,1},{2,3},{4,5},{6,7},{8,9},{10,11}};
    for (int layer = 0; layer < 2; ++layer) {
        for (int k = 0; k < 12; ++k) {          // add_U(i,j)
            int i = UP[k][0], j = UP[k][1];
            B.u3(i); B.u3(j);
            B.cnot(j, i);
            B.rz(i); B.ry(j);
            B.cnot(i, j);
            B.ry(j);
            B.cnot(j, i);
            B.u3(i); B.u3(j);
        }
        for (int k = 0; k < 6; ++k) {           // add_V(i,j)
            int i = VP[k][0], j = VP[k][1];
            B.u3(i);
            int gp = B.p;                       // params of G
            B.u3(j);                            // G on j
            B.cnot(i, j);
            B.u3dag(j, gp);                     // G^dagger on j
        }
    }
    // add_F over all 12 wires
    for (int i = 0; i < 12; ++i) B.u3(i);
    for (int i = 0; i < 12; ++i) { int j = (i + 1) % 12; B.cnot(i, j); B.cnot(j, i); }
    for (int i = 0; i < 12; ++i) B.u3(i);
    for (int i = 0; i < 12; ++i) { int j = (i + 3) % 12; B.cnot(i, j); B.cnot(j, i); }
    return B;
}

constexpr Builder kBuilt = make_builder();
static_assert(kBuilt.n == NOPS, "op count mismatch");
static_assert(kBuilt.p == 504,  "param count mismatch");
__constant__ OpTable c_ops = kBuilt.T;

// ---------------------------------------------------------------------------
// Kernel 1: per-state circuit simulation in LDS + expectations -> fake (f16)
// one workgroup (256 threads = 8 wave32) per batch element
// ---------------------------------------------------------------------------
__global__ __launch_bounds__(256)
void qnn_sim_kernel(const float* __restrict__ zr, const float* __restrict__ zi,
                    const float* __restrict__ params, _Float16* __restrict__ fakeh) {
    __shared__ float sre[DIM];
    __shared__ float simg[DIM];
    __shared__ float ex[NEXP];
    __shared__ float rnorm_s;

    const int tid = threadIdx.x;
    const int b   = blockIdx.x;

    // load state into LDS
    for (int k = tid; k < DIM; k += 256) {
        sre[k]  = zr[(size_t)b * DIM + k];
        simg[k] = zi[(size_t)b * DIM + k];
    }
    __syncthreads();

    // gate loop
    for (int g = 0; g < NOPS; ++g) {
        const Op op = c_ops.ops[g];
        if (op.kind == 4) {                         // CNOT: swap target pair where ctrl==1
            const int cb = 11 - op.a, tb = 11 - op.b;
            const unsigned mask = (1u << tb) - 1u;
            for (int k = tid; k < NPAIRS; k += 256) {
                unsigned i0 = (((unsigned)k & ~mask) << 1) | ((unsigned)k & mask);
                if ((i0 >> cb) & 1u) {
                    unsigned i1 = i0 | (1u << tb);
                    float tr = sre[i0];  sre[i0]  = sre[i1];  sre[i1]  = tr;
                    float ti = simg[i0]; simg[i0] = simg[i1]; simg[i1] = ti;
                }
            }
        } else {                                    // general 2x2 complex unitary
            float u00r, u00i, u01r, u01i, u10r, u10i, u11r, u11i;
            const float* P = params + op.p;
            if (op.kind == 0 || op.kind == 1) {     // U3 / U3^dagger
                const float th = P[0], ph = P[1], la = P[2];
                const float ct = cosf(0.5f * th), st = sinf(0.5f * th);
                const float cl = cosf(la), sl = sinf(la);
                const float cp = cosf(ph), sp = sinf(ph);
                const float cpl = cosf(ph + la), spl = sinf(ph + la);
                if (op.kind == 0) {
                    u00r = ct;        u00i = 0.f;
                    u01r = -cl * st;  u01i = -sl * st;
                    u10r =  cp * st;  u10i =  sp * st;
                    u11r =  cpl * ct; u11i =  spl * ct;
                } else {                            // conj transpose of U3
                    u00r = ct;        u00i = 0.f;
                    u01r =  cp * st;  u01i = -sp * st;
                    u10r = -cl * st;  u10i =  sl * st;
                    u11r =  cpl * ct; u11i = -spl * ct;
                }
            } else if (op.kind == 2) {              // RZ
                const float th = P[0];
                const float c = cosf(0.5f * th), s = sinf(0.5f * th);
                u00r = c;   u00i = -s;
                u01r = 0.f; u01i = 0.f;
                u10r = 0.f; u10i = 0.f;
                u11r = c;   u11i =  s;
            } else {                                // RY
                const float th = P[0];
                const float c = cosf(0.5f * th), s = sinf(0.5f * th);
                u00r = c;  u00i = 0.f;
                u01r = -s; u01i = 0.f;
                u10r = s;  u10i = 0.f;
                u11r = c;  u11i = 0.f;
            }
            const int tb = 11 - op.a;
            const unsigned mask = (1u << tb) - 1u;
            for (int k = tid; k < NPAIRS; k += 256) {
                unsigned i0 = (((unsigned)k & ~mask) << 1) | ((unsigned)k & mask);
                unsigned i1 = i0 | (1u << tb);
                float r0 = sre[i0], q0 = simg[i0];
                float r1 = sre[i1], q1 = simg[i1];
                float n0r = u00r * r0 - u00i * q0 + u01r * r1 - u01i * q1;
                float n0i = u00r * q0 + u00i * r0 + u01r * q1 + u01i * r1;
                float n1r = u10r * r0 - u10i * q0 + u11r * r1 - u11i * q1;
                float n1i = u10r * q0 + u10i * r0 + u11r * q1 + u11i * r1;
                sre[i0] = n0r; simg[i0] = n0i;
                sre[i1] = n1r; simg[i1] = n1i;
            }
        }
        __syncthreads();
    }

    // Pauli expectations: exps[q*3+{X,Y,Z}]
    if (tid < NEXP) ex[tid] = 0.f;
    __syncthreads();
    for (int q = 0; q < NQ; ++q) {
        const int tb = 11 - q;
        const unsigned mask = (1u << tb) - 1u;
        float sx = 0.f, sy = 0.f, sz = 0.f;
        for (int k = tid; k < NPAIRS; k += 256) {
            unsigned i0 = (((unsigned)k & ~mask) << 1) | ((unsigned)k & mask);
            unsigned i1 = i0 | (1u << tb);
            float r0 = sre[i0], q0 = simg[i0];
            float r1 = sre[i1], q1 = simg[i1];
            sx += 2.f * (r0 * r1 + q0 * q1);               // 2*Re(conj(a0)*a1)
            sy += 2.f * (r0 * q1 - q0 * r1);               // 2*Im(conj(a0)*a1)
            sz += (r0 * r0 + q0 * q0) - (r1 * r1 + q1 * q1);
        }
        atomicAdd(&ex[3 * q + 0], sx);
        atomicAdd(&ex[3 * q + 1], sy);
        atomicAdd(&ex[3 * q + 2], sz);
    }
    __syncthreads();
    if (tid == 0) {
        float s = 0.f;
        for (int k = 0; k < NEXP; ++k) s += ex[k] * ex[k];
        rnorm_s = rsqrtf(s);
    }
    __syncthreads();
    if (tid < KPAD) {
        float v = (tid < NEXP) ? ex[tid] * rnorm_s : 0.f;
        fakeh[(size_t)b * KPAD + tid] = (_Float16)v;
    }
}

// ---------------------------------------------------------------------------
// Kernel 2: fid = fake @ fake^T via WMMA f32_16x16x32_f16, masked sums
// one wave32 per 16x16 output tile; 8 waves per block; 128x128 tiles total
// ---------------------------------------------------------------------------
typedef __attribute__((ext_vector_type(16))) _Float16 v16h;
typedef __attribute__((ext_vector_type(8)))  _Float16 v8h;
typedef __attribute__((ext_vector_type(8)))  float    v8f;

__global__ __launch_bounds__(256)
void qnn_fid_kernel(const _Float16* __restrict__ F, const int* __restrict__ y,
                    float* __restrict__ accum) {
    const int tid  = threadIdx.x;
    const int wid  = tid >> 5;
    const int lane = tid & 31;
    const int tile = blockIdx.x * 8 + wid;
    const int tm   = tile >> 7;          // 128 tiles per row
    const int tn   = tile & 127;
    const int mrow = lane & 15;
    const int h    = lane >> 4;

    const _Float16* Arow = F + (size_t)(tm * 16 + mrow) * KPAD;  // A: 16xK rows
    const _Float16* Brow = F + (size_t)(tn * 16 + mrow) * KPAD;  // B cols = fake rows

    v16h a0, a1, b0, b1;
    a0.lo = *(const v8h*)(Arow + 8 * h);
    a0.hi = *(const v8h*)(Arow + 16 + 8 * h);
    a1.lo = *(const v8h*)(Arow + 32 + 8 * h);
    a1.hi = *(const v8h*)(Arow + 48 + 8 * h);
    b0.lo = *(const v8h*)(Brow + 8 * h);
    b0.hi = *(const v8h*)(Brow + 16 + 8 * h);
    b1.lo = *(const v8h*)(Brow + 32 + 8 * h);
    b1.hi = *(const v8h*)(Brow + 48 + 8 * h);

    v8f c = {};
    c = __builtin_amdgcn_wmma_f32_16x16x32_f16(false, a0, false, b0, (short)0, c, false, false);
    c = __builtin_amdgcn_wmma_f32_16x16x32_f16(false, a1, false, b1, (short)0, c, false, false);

    // D layout: lane n = lane&15; VGPR r -> row m = 8*(lane>>4)+r
    const int gj = tn * 16 + (lane & 15);
    const int yj = y[gj];
    float s_tot = 0.f, s_eq = 0.f;
#pragma unroll
    for (int r = 0; r < 8; ++r) {
        const int gi = tm * 16 + 8 * h + r;
        const float v = c[r];
        s_tot += v;
        if (y[gi] == yj) s_eq += v;
    }

    __shared__ float red[2];
    if (tid < 2) red[tid] = 0.f;
    __syncthreads();
    atomicAdd(&red[0], s_eq);
    atomicAdd(&red[1], s_tot);
    __syncthreads();
    if (tid == 0) {
        atomicAdd(&accum[0], red[0]);
        atomicAdd(&accum[1], red[1]);
    }
}

// ---------------------------------------------------------------------------
// Kernel 0: zero accumulators      Kernel 3: finalize 3 scalars
// ---------------------------------------------------------------------------
__global__ void qnn_zero_kernel(float* accum) {
    if (threadIdx.x < 2) accum[threadIdx.x] = 0.f;
}

__global__ __launch_bounds__(256)
void qnn_final_kernel(const int* __restrict__ y, const float* __restrict__ accum,
                      float* __restrict__ out) {
    __shared__ int cnt[NCLS];
    const int tid = threadIdx.x;
    if (tid < NCLS) cnt[tid] = 0;
    __syncthreads();
    for (int i = tid; i < BATCH; i += 256) atomicAdd(&cnt[y[i]], 1);
    __syncthreads();
    if (tid == 0) {
        float sum_eq = 0.f;
        for (int c = 0; c < NCLS; ++c) sum_eq += (float)cnt[c] * (float)cnt[c];
        const float total = (float)BATCH * (float)BATCH;
        const float S_eq = accum[0], S_tot = accum[1];
        const float fid_eq = S_eq / sum_eq;
        const float fid_ne = (S_tot - S_eq) / (total - sum_eq);
        out[0] = fid_ne - fid_eq;
        out[1] = fid_eq;
        out[2] = fid_ne;
    }
}

// ---------------------------------------------------------------------------
// Launcher
// ---------------------------------------------------------------------------
extern "C" void kernel_launch(void* const* d_in, const int* in_sizes, int n_in,
                              void* d_out, int out_size, void* d_ws, size_t ws_size,
                              hipStream_t stream) {
    (void)in_sizes; (void)n_in; (void)out_size; (void)ws_size;
    const float* zr     = (const float*)d_in[0];
    const float* zi     = (const float*)d_in[1];
    const float* params = (const float*)d_in[2];
    const int*   y      = (const int*)d_in[3];
    float*       out    = (float*)d_out;

    _Float16* fakeh = (_Float16*)d_ws;                                   // 2048*64 f16 = 256 KB
    float*    accum = (float*)((char*)d_ws + (size_t)BATCH * KPAD * 2);  // 2 floats

    qnn_zero_kernel<<<1, 32, 0, stream>>>(accum);
    qnn_sim_kernel<<<BATCH, 256, 0, stream>>>(zr, zi, params, fakeh);
    qnn_fid_kernel<<<(128 * 128) / 8, 256, 0, stream>>>(fakeh, y, accum);
    qnn_final_kernel<<<1, 256, 0, stream>>>(y, accum, out);
}
